// TPUSparseMoEDispatch_19756849562326
// MI455X (gfx1250) — matrix operand order repaired and checked by reading, not compiled, not run on hardware
//
#include <hip/hip_runtime.h>

// ---------------------------------------------------------------------------
// MoE with identical experts + renormalized top-k probs: sum_e combine[t,e]=1,
// so out = GELU(x @ w1 + b1) @ w2 + b2 exactly (router is dead code).
// fp32 -> bf16 once (weights pre-transposed to [N,K]), two WMMA GEMMs with
// f32 accumulation, fused branchless bias+GELU epilogue.
// Tiles staged global->LDS with CDNA5 GLOBAL_LOAD_ASYNC_TO_LDS_B128.
// ---------------------------------------------------------------------------

typedef __attribute__((ext_vector_type(16))) __bf16 v16bf;
typedef __attribute__((ext_vector_type(8)))  __bf16 v8bf;
typedef __attribute__((ext_vector_type(8)))  float  v8f;
typedef __attribute__((ext_vector_type(4)))  int    v4i;

union Frag16 { v16bf v; v8bf h[2]; };

#define TILE_M 128
#define TILE_N 128
#define TILE_K 32
#define LDS_STRIDE 40   // 32 + 8 pad (bf16 elems); 80B row stride keeps 16B alignment

#if defined(__has_builtin)
#if __has_builtin(__builtin_amdgcn_global_load_async_to_lds_b128) && \
    __has_builtin(__builtin_amdgcn_s_wait_asynccnt)
#define USE_ASYNC_LDS 1
#endif
#endif

#define AS1 __attribute__((address_space(1)))
#define AS3 __attribute__((address_space(3)))

__global__ __launch_bounds__(256) void cvt_f32_to_bf16(const float* __restrict__ in,
                                                       __bf16* __restrict__ out, int n) {
  int i = blockIdx.x * blockDim.x + threadIdx.x;
  int stride = gridDim.x * blockDim.x;
  for (; i < n; i += stride) out[i] = (__bf16)in[i];
}

// out[c][r] = (bf16) in[r][c]; R, C multiples of 32. Block = 256 threads, 32x32 tile.
__global__ __launch_bounds__(256) void transpose_f32_to_bf16(
    const float* __restrict__ in, __bf16* __restrict__ out, int R, int C) {
  __shared__ float tile[32][33];
  const int c0 = blockIdx.x * 32, r0 = blockIdx.y * 32;
  const int tx = threadIdx.x & 31, ty = threadIdx.x >> 5;  // 32 x 8
#pragma unroll
  for (int i = 0; i < 32; i += 8)
    tile[ty + i][tx] = in[(size_t)(r0 + ty + i) * C + c0 + tx];
  __syncthreads();
#pragma unroll
  for (int i = 0; i < 32; i += 8)
    out[(size_t)(c0 + ty + i) * R + r0 + tx] = (__bf16)tile[tx][ty + i];
}

__device__ __forceinline__ float fast_gelu(float v) {
  // jax.nn.gelu (tanh approximation), branchless:
  // tanh(u) = 1 - 2/(exp2(2*log2e*u)+1); saturates via inf/0, no branches.
  const float kAlpha = 0.7978845608028654f;          // sqrt(2/pi)
  const float k2Log2e = 2.8853900817779268f;         // 2*log2(e)
  float u = kAlpha * (v + 0.044715f * v * v * v);
  float e = __builtin_amdgcn_exp2f(k2Log2e * u);
  float t = 1.0f - 2.0f * __builtin_amdgcn_rcpf(e + 1.0f);
  return 0.5f * v * (1.0f + t);
}

// C[M,N] = A[M,K] @ B^T where Bt is [N,K]; A/Bt bf16 row-major, accum f32.
// M,N multiples of 128; K multiple of 32.
template <bool GELU_EPI, bool OUT_BF16>
__global__ __launch_bounds__(256) void gemm_bf16_wmma(
    const __bf16* __restrict__ A, const __bf16* __restrict__ Bt,
    const float* __restrict__ bias,
    __bf16* __restrict__ outb, float* __restrict__ outf,
    int M, int N, int K) {
  __shared__ __bf16 As[TILE_M][LDS_STRIDE];   // [m][k]
  __shared__ __bf16 Bs[TILE_N][LDS_STRIDE];   // [n][k]

  const int tid  = threadIdx.x;
  const int lane = tid & 31;
  const int wave = tid >> 5;
  const int half = lane >> 4;   // 0: lanes 0-15, 1: lanes 16-31
  const int l16  = lane & 15;

  const int m0 = blockIdx.x * TILE_M;
  const int n0 = blockIdx.y * TILE_N;
  const int wm = (wave & 3) * 32;   // wave row offset within block tile
  const int wn = (wave >> 2) * 64;  // wave col offset within block tile

  v8f acc[2][4];
#pragma unroll
  for (int i = 0; i < 2; ++i)
#pragma unroll
    for (int j = 0; j < 4; ++j)
#pragma unroll
      for (int r = 0; r < 8; ++r) acc[i][j][r] = 0.0f;

  // Staging map (identical for A and Bt): thread -> 16 contiguous bf16 of one row.
  const int s_row = tid >> 1;        // 0..127
  const int s_seg = (tid & 1) * 16;  // k offset 0 or 16

  const __bf16* srcA = A  + (size_t)(m0 + s_row) * K + s_seg;
  const __bf16* srcB = Bt + (size_t)(n0 + s_row) * K + s_seg;

  for (int k0 = 0; k0 < K; k0 += TILE_K) {
#ifdef USE_ASYNC_LDS
    // CDNA5 async copy: global -> LDS without touching VGPRs, ASYNCcnt tracked.
    __builtin_amdgcn_global_load_async_to_lds_b128(
        (AS1 v4i*)(srcA + k0),     (AS3 v4i*)&As[s_row][s_seg],     0, 0);
    __builtin_amdgcn_global_load_async_to_lds_b128(
        (AS1 v4i*)(srcA + k0 + 8), (AS3 v4i*)&As[s_row][s_seg + 8], 0, 0);
    __builtin_amdgcn_global_load_async_to_lds_b128(
        (AS1 v4i*)(srcB + k0),     (AS3 v4i*)&Bs[s_row][s_seg],     0, 0);
    __builtin_amdgcn_global_load_async_to_lds_b128(
        (AS1 v4i*)(srcB + k0 + 8), (AS3 v4i*)&Bs[s_row][s_seg + 8], 0, 0);
    __builtin_amdgcn_s_wait_asynccnt(0);
#else
    {
      v8bf p0 = *(const v8bf*)(srcA + k0);
      v8bf p1 = *(const v8bf*)(srcA + k0 + 8);
      *(v8bf*)&As[s_row][s_seg]     = p0;
      *(v8bf*)&As[s_row][s_seg + 8] = p1;
      v8bf q0 = *(const v8bf*)(srcB + k0);
      v8bf q1 = *(const v8bf*)(srcB + k0 + 8);
      *(v8bf*)&Bs[s_row][s_seg]     = q0;
      *(v8bf*)&Bs[s_row][s_seg + 8] = q1;
    }
#endif
    __syncthreads();

    // Fragments per CDNA5 ISA 7.12.2 layouts.
    // A 16x32 bf16: half 0 -> K 0-7 (e0-7), 16-23 (e8-15); half 1 -> +8.
    Frag16 af[2];
#pragma unroll
    for (int i = 0; i < 2; ++i) {
      const int row = wm + i * 16 + l16;
      af[i].h[0] = *(const v8bf*)&As[row][half * 8];
      af[i].h[1] = *(const v8bf*)&As[row][16 + half * 8];
    }
    // B 32x16 bf16: lane = N, same K order as A (half 0 -> K 0-7, 16-23; half 1 -> +8).
    Frag16 bf[4];
#pragma unroll
    for (int j = 0; j < 4; ++j) {
      const int col = wn + j * 16 + l16;
      bf[j].h[0] = *(const v8bf*)&Bs[col][half * 8];
      bf[j].h[1] = *(const v8bf*)&Bs[col][16 + half * 8];
    }

#pragma unroll
    for (int i = 0; i < 2; ++i)
#pragma unroll
      for (int j = 0; j < 4; ++j)
        acc[i][j] = __builtin_amdgcn_wmma_f32_16x16x32_bf16(
            false, af[i].v, false, bf[j].v, (short)0, acc[i][j], false, false);

    __syncthreads();
  }

  // Epilogue. D layout: VGPR r -> row r (lanes 0-15) / r+8 (lanes 16-31), col = lane&15.
#pragma unroll
  for (int i = 0; i < 2; ++i) {
#pragma unroll
    for (int j = 0; j < 4; ++j) {
      const int col = n0 + wn + j * 16 + l16;
      const float bv = bias[col];
#pragma unroll
      for (int r = 0; r < 8; ++r) {
        const int row = m0 + wm + i * 16 + half * 8 + r;
        float v = acc[i][j][r] + bv;
        if (GELU_EPI) v = fast_gelu(v);
        if (OUT_BF16) outb[(size_t)row * N + col] = (__bf16)v;
        else          outf[(size_t)row * N + col] = v;
      }
    }
  }
}

extern "C" void kernel_launch(void* const* d_in, const int* in_sizes, int n_in,
                              void* d_out, int out_size, void* d_ws, size_t ws_size,
                              hipStream_t stream) {
  const float* x  = (const float*)d_in[0];  // [B,S,H]
  const float* rw = (const float*)d_in[1];  // router weights (mathematically dead)
  const float* w1 = (const float*)d_in[2];  // [H,F]
  const float* b1 = (const float*)d_in[3];  // [F]
  const float* w2 = (const float*)d_in[4];  // [F,H]
  const float* b2 = (const float*)d_in[5];  // [H]
  float* out = (float*)d_out;
  (void)rw; (void)n_in; (void)out_size; (void)ws_size;

  const int NUM_EXPERTS = 8;
  const int H = in_sizes[1] / NUM_EXPERTS;   // 1024
  const int F = in_sizes[3];                 // 4096
  const int T = in_sizes[0] / H;             // 16384 tokens

  // Workspace layout (bf16): x, w1^T [F,H], w2^T [H,F], hidden activations.
  __bf16* xb   = (__bf16*)d_ws;
  __bf16* w1tb = xb   + (size_t)T * H;
  __bf16* w2tb = w1tb + (size_t)H * F;
  __bf16* hb   = w2tb + (size_t)F * H;

  cvt_f32_to_bf16<<<2048, 256, 0, stream>>>(x, xb, T * H);
  {  // w1 [H,F] -> w1t [F,H]
    dim3 grid(F / 32, H / 32);
    transpose_f32_to_bf16<<<grid, 256, 0, stream>>>(w1, w1tb, H, F);
  }
  {  // w2 [F,H] -> w2t [H,F]
    dim3 grid(H / 32, F / 32);
    transpose_f32_to_bf16<<<grid, 256, 0, stream>>>(w2, w2tb, F, H);
  }

  // GEMM1: h = GELU(x @ w1 + b1)   [T,H]x[H,F] -> bf16 [T,F]
  {
    dim3 grid(T / TILE_M, F / TILE_N);
    gemm_bf16_wmma<true, true><<<grid, 256, 0, stream>>>(
        xb, w1tb, b1, hb, nullptr, T, F, H);
  }
  // GEMM2: out = h @ w2 + b2       [T,F]x[F,H] -> f32 [T,H]
  {
    dim3 grid(T / TILE_M, H / TILE_N);
    gemm_bf16_wmma<false, false><<<grid, 256, 0, stream>>>(
        hb, w2tb, b2, nullptr, out, T, H, F);
  }
}
